// Equilibrium_Propagation_Reward_Policy_Network_62139586839108
// MI455X (gfx1250) — matrix operand order repaired and checked by reading, not compiled, not run on hardware
//
#include <hip/hip_runtime.h>

typedef _Float16 half_t;
typedef __attribute__((ext_vector_type(16))) _Float16 v16h;
typedef __attribute__((ext_vector_type(8)))  _Float16 v8h;
typedef __attribute__((ext_vector_type(8)))  float    v8f;

#define HID   128
#define OUTN  4096
#define NIN   65536
#define NBLK  16          // cooperating blocks; each owns 256 output columns
#define COLS  256
#define TPB   256         // 8 waves of 32
#define W2C_S 136         // halfs; 272B row stride -> 68 dwords = 4 mod 64 banks
#define W2R_S 264         // halfs; 528B row stride -> 132 dwords = 4 mod 64 banks

#define CAT16(lo, hi8) __builtin_shufflevector(lo, hi8, 0,1,2,3,4,5,6,7,8,9,10,11,12,13,14,15)

__device__ __forceinline__ float clip01(float v) { return fminf(fmaxf(v, 0.f), 1.f); }
__device__ __forceinline__ float drho(float v)   { return (v >= 0.f && v <= 1.f) ? 1.f : 0.f; }

// ---------------- init: zero accumulator + barrier state in workspace ----------------
__global__ void eqprop_init(float* xw1, unsigned* ctr, unsigned* gen) {
  int t = threadIdx.x;
  if (t < HID) xw1[t] = 0.f;
  if (t == 128) *ctr = 0u;
  if (t == 129) *gen = 0u;
}

// ---------------- phase 1: xW1 = clip(x,0,1) @ W1  (streams 32MB of W1) ----------------
__global__ void eqprop_xw1(const float* __restrict__ x, const float* __restrict__ W1,
                           float* __restrict__ xw1) {
  int c = threadIdx.x & (HID - 1);   // column 0..127
  int g = threadIdx.x >> 7;          // row-group 0/1
  float acc = 0.f;
  for (int i = blockIdx.x * 2 + g; i < NIN; i += gridDim.x * 2)
    acc = fmaf(clip01(x[i]), W1[i * HID + c], acc);
  __shared__ float s[HID];
  if (g == 0) s[c] = acc;
  __syncthreads();
  if (g == 1) atomicAdd(&xw1[c], acc + s[c]);
}

// ---------------- phase 2: persistent cooperative fixed-point loop ----------------
__global__ void __launch_bounds__(TPB, 1)
eqprop_main(const float* __restrict__ W2, const float* __restrict__ b_h,
            const float* __restrict__ b_out, const float* __restrict__ hidden0,
            const float* __restrict__ output0, const int* __restrict__ n_iter_p,
            const float* __restrict__ eps_p, const float* __restrict__ xw1,
            float* __restrict__ gpart, unsigned* __restrict__ ctr,
            unsigned* __restrict__ gen, float* __restrict__ out) {
  // W2 slice resident in LDS for the whole loop, in both majors (f16):
  __shared__ __attribute__((aligned(16))) half_t W2c[COLS * W2C_S]; // [j][k] for B operand
  __shared__ __attribute__((aligned(16))) half_t W2r[HID  * W2R_S]; // [k][j] for A operand
  // activation vectors with a same-size permanently-zero shadow region:
  // inactive lanes read the shadow -> unconditional vector LDS loads, no predication
  __shared__ __attribute__((aligned(16))) half_t rh16[2 * HID];
  __shared__ __attribute__((aligned(16))) half_t ro16[2 * COLS];
  __shared__ float go_buf[COLS];
  __shared__ float p_buf[HID];

  const int tid = threadIdx.x;
  const int nb  = blockIdx.x;
  const int w   = tid >> 5;     // wave id 0..7
  const int l   = tid & 31;     // lane
  const int lm  = l & 15;
  const bool hi = (l >= 16);
  const int abase = (lm == 0) ? 0 : HID;   // rh16 base: live region vs zero region
  const int bbase = (lm == 0) ? 0 : COLS;  // ro16 base: live region vs zero region

  // Stage W2 columns [nb*256, nb*256+256) -> LDS (coalesced global reads).
  for (int idx = tid; idx < HID * COLS; idx += TPB) {
    int k = idx >> 8;
    int j = idx & (COLS - 1);
    half_t hv = (half_t)W2[k * OUTN + nb * COLS + j];
    W2c[j * W2C_S + k] = hv;
    W2r[k * W2R_S + j] = hv;
  }
  // zero the shadow regions once
  if (tid < HID) rh16[HID + tid] = (half_t)0.f;
  ro16[COLS + tid] = (half_t)0.f;

  const int col = nb * COLS + tid;      // this thread's output unit
  float o  = output0[col];
  float bo = b_out[col];
  float mo = 0.f, vo = 0.f;
  float h = 0.f, bh = 0.f, xk = 0.f, mh = 0.f, vh = 0.f;
  if (tid < HID) { h = hidden0[tid]; bh = b_h[tid]; xk = xw1[tid]; }

  const int   niter = n_iter_p[0];
  const float eps   = eps_p[0];
  float p1 = 1.f, p2 = 1.f;

  for (int t = 0; t < niter; ++t) {
    if (tid < HID) rh16[tid] = (half_t)clip01(h);
    ro16[tid] = (half_t)clip01(o);
    __syncthreads();

    // ---- g_o dots: D[0][n] = sum_k r_h[k] * W2[k][n]  (wave w -> n-tiles 2w, 2w+1)
    {
      v8f c0 = {0.f, 0.f, 0.f, 0.f, 0.f, 0.f, 0.f, 0.f};
      v8f c1 = {0.f, 0.f, 0.f, 0.f, 0.f, 0.f, 0.f, 0.f};
      #pragma unroll
      for (int kb = 0; kb < 4; ++kb) {
        // A (16x32 f16): row 0 = r_h segment; lanes 0/16 carry the two K halves,
        // all other lanes fetch from the zero shadow region (same addresses +HID).
        const int kbase = abase + kb * 32 + (hi ? 8 : 0);
        v8h a0 = *(const v8h*)&rh16[kbase];
        v8h a1 = *(const v8h*)&rh16[kbase + 16];
        v16h a = CAT16(a0, a1);
        // B (32x16 f16): lane -> col nt*16+lm, K = 8*hi + i (i<8), 16 + 8*hi + (i-8) (i>=8)
        const int bofs = lm * W2C_S + kb * 32 + (hi ? 8 : 0);
        const half_t* bp0 = &W2c[(2 * w) * 16 * W2C_S + bofs];
        const half_t* bp1 = &W2c[(2 * w + 1) * 16 * W2C_S + bofs];
        v16h b0 = CAT16(*(const v8h*)bp0, *(const v8h*)(bp0 + 16));
        v16h b1 = CAT16(*(const v8h*)bp1, *(const v8h*)(bp1 + 16));
        c0 = __builtin_amdgcn_wmma_f32_16x16x32_f16(false, a, false, b0, (short)0, c0, false, false);
        c1 = __builtin_amdgcn_wmma_f32_16x16x32_f16(false, a, false, b1, (short)0, c1, false, false);
      }
      if (!hi) {                               // D row 0 lives in VGPR0, lanes 0..15
        go_buf[(2 * w) * 16 + lm]     = c0[0];
        go_buf[(2 * w + 1) * 16 + lm] = c1[0];
      }
    }

    // ---- g_h partials: p[m] = sum_j W2[m][j] * r_o[j]  (wave w -> m-tile 16w..16w+15)
    {
      v8f c = {0.f, 0.f, 0.f, 0.f, 0.f, 0.f, 0.f, 0.f};
      #pragma unroll
      for (int jb = 0; jb < 8; ++jb) {
        // A (16x32 f16): row m = 16w+lm from the row-major copy
        const half_t* ap = &W2r[(16 * w + lm) * W2R_S + jb * 32 + (hi ? 8 : 0)];
        v16h a = CAT16(*(const v8h*)ap, *(const v8h*)(ap + 16));
        // B (32x16 f16): only column N=0 nonzero (r_o segment); inactive lanes
        // read the zero shadow region
        const int jbase = bbase + jb * 32 + (hi ? 8 : 0);
        v8h q0 = *(const v8h*)&ro16[jbase];
        v8h q1 = *(const v8h*)&ro16[jbase + 16];
        v16h b = CAT16(q0, q1);
        c = __builtin_amdgcn_wmma_f32_16x16x32_f16(false, a, false, b, (short)0, c, false, false);
      }
      if (lm == 0) {                     // D col 0: lane0 -> M=r, lane16 -> M=r+8
        const int mb = 16 * w + (hi ? 8 : 0);
        #pragma unroll
        for (int r = 0; r < 8; ++r) p_buf[mb + r] = c[r];
      }
    }
    __syncthreads();

    // publish this block's g_h partial (double-buffered by iteration parity)
    if (tid < HID) gpart[(t & 1) * NBLK * HID + nb * HID + tid] = p_buf[tid];

    p1 *= 0.9f; p2 *= 0.999f;
    const float c1r = 1.f / (1.f - p1);
    const float c2r = 1.f / (1.f - p2);

    { // Adam on owned output unit (purely block-local)
      const float ro = clip01(o);
      const float go = drho(o) * (ro - bo - go_buf[tid]);
      mo = 0.9f  * mo + 0.1f   * go;
      vo = 0.999f * vo + 0.001f * go * go;
      o -= eps * (mo * c1r) / (sqrtf(vo * c2r) + 1e-8f);
    }

    // ---- one device-scope barrier per iteration across the 16 blocks ----
    __threadfence();
    __syncthreads();
    if (tid == 0) {
      unsigned a = __hip_atomic_fetch_add(ctr, 1u, __ATOMIC_ACQ_REL, __HIP_MEMORY_SCOPE_AGENT);
      if (a == NBLK - 1) {
        __hip_atomic_store(ctr, 0u, __ATOMIC_RELAXED, __HIP_MEMORY_SCOPE_AGENT);
        __hip_atomic_store(gen, (unsigned)(t + 1), __ATOMIC_RELEASE, __HIP_MEMORY_SCOPE_AGENT);
      } else {
        while (__hip_atomic_load(gen, __ATOMIC_ACQUIRE, __HIP_MEMORY_SCOPE_AGENT) < (unsigned)(t + 1))
          __builtin_amdgcn_s_sleep(2);
      }
    }
    __syncthreads();

    if (tid < HID) { // Adam on hidden unit, replicated identically in all blocks
      float s = 0.f;
      #pragma unroll
      for (int b = 0; b < NBLK; ++b)
        s += __hip_atomic_load(&gpart[(t & 1) * NBLK * HID + b * HID + tid],
                               __ATOMIC_RELAXED, __HIP_MEMORY_SCOPE_AGENT);
      const float rh = clip01(h);
      const float gh = drho(h) * (rh - bh - xk - s);
      mh = 0.9f  * mh + 0.1f   * gh;
      vh = 0.999f * vh + 0.001f * gh * gh;
      h -= eps * (mh * c1r) / (sqrtf(vh * c2r) + 1e-8f);
    }
  }

  out[col] = o;
}

extern "C" void kernel_launch(void* const* d_in, const int* in_sizes, int n_in,
                              void* d_out, int out_size, void* d_ws, size_t ws_size,
                              hipStream_t stream) {
  const float* x   = (const float*)d_in[0];
  const float* W1  = (const float*)d_in[1];
  const float* W2  = (const float*)d_in[2];
  // d_in[3] = b_in (unused by the reference output)
  const float* bh  = (const float*)d_in[4];
  const float* bo  = (const float*)d_in[5];
  const float* h0  = (const float*)d_in[6];
  const float* o0  = (const float*)d_in[7];
  const int*   nit = (const int*)d_in[8];
  const float* eps = (const float*)d_in[9];
  float* out = (float*)d_out;

  float*    xw1   = (float*)d_ws;                       // 128 f32
  float*    gpart = xw1 + HID;                          // 2 * 16 * 128 f32
  unsigned* ctr   = (unsigned*)(gpart + 2 * NBLK * HID);
  unsigned* gen   = ctr + 1;

  eqprop_init<<<1, 256, 0, stream>>>(xw1, ctr, gen);
  eqprop_xw1 <<<512, 256, 0, stream>>>(x, W1, xw1);
  eqprop_main<<<NBLK, TPB, 0, stream>>>(W2, bh, bo, h0, o0, nit, eps,
                                        xw1, gpart, ctr, gen, out);
}
